// ACaWMSA_11261404250483
// MI455X (gfx1250) — matrix operand order repaired and checked by reading, not compiled, not run on hardware
//
#include <hip/hip_runtime.h>
#include <hip/hip_bf16.h>

// ACaWMSA block for MI455X (gfx1250, wave32, WMMA + TDM).
// All dense matmuls (p1 1x1conv, qk/v 1x1convs, proj, fc1, fc2) and both
// attention matmuls (q@k^T, P@v) go through v_wmma_f32_16x16x32_f16.
// GEMM weight tiles are staged into LDS by the Tensor Data Mover
// (tensor_load_to_lds + s_wait_tensorcnt; 6-arg builtin on this toolchain).
// Intermediates are f16 (bandwidth-bound problem @ 23.3 TB/s HBM).
//
// d_in flattening assumed (setup_inputs dict order, recursive):
//  0:x 1:n1_g 2:n1_b 3:n2_g 4:n2_b 5:p1_w 6:p1_b 7:proj_w 8:proj_b
//  9..18:a1{qk_w,qk_b,q5_w,q5_b,k5_w,k5_b,v_w,v_b,cape_w,cape_b} 19..28:a2
//  29..38:a3 39..48:a4 49:fc1_w 50:fc1_b 51:fc2_w 52:fc2_b
//  53:dw1_w 54:dw1_b 55:dw2_w 56:dw2_b 57:dw3_w 58:dw3_b
//  59:dw4a_w 60:dw4a_b 61:dw4b_w 62:dw4b_b

typedef _Float16 f16;
typedef __attribute__((ext_vector_type(16))) _Float16 v16h;
typedef __attribute__((ext_vector_type(8)))  float    v8f;
typedef __attribute__((ext_vector_type(4)))  unsigned v4u;
typedef __attribute__((ext_vector_type(8)))  int      v8i_t;
typedef __attribute__((ext_vector_type(4)))  int      v4i_t;

#if __has_builtin(__builtin_amdgcn_tensor_load_to_lds)
#define HAVE_TDM 1
#else
#define HAVE_TDM 0
#endif

#define HH 256
#define WW 256
#define BB 4
#define CDIM 128
#define NPIX (BB*HH*WW)      // 262144
#define SHIFT_ 4
#define SCALE_Q 0.17677669529663687f   // 32^-0.5
#define VSTRIDE 72           // padded f16 stride for transposed V in LDS

__device__ __forceinline__ float waveSum(float v) {
  #pragma unroll
  for (int off = 16; off; off >>= 1) v += __shfl_xor(v, off, 32);
  return v;
}

// K-offset (in f16 elems) for 16-bit A-matrix fragment, VGPR j, lane half.
__device__ __forceinline__ int a_koff(int j, int half) {
  return (j << 1) + (half << 4);
}

// ---------------------------------------------------------------- weights f32->f16
__global__ void packw_kernel(const float* __restrict__ src, f16* __restrict__ dst, int n) {
  int i = blockIdx.x * 256 + threadIdx.x;
  if (i < n) dst[i] = (f16)src[i];
}

// ---------------------------------------------------------------- LN1: roll(-4,-4) + LayerNorm
// x: NCHW f32 -> out: [N,128] f16 (rolled frame). One wave per pixel.
__global__ __launch_bounds__(256) void ln_shift_kernel(
    const float* __restrict__ x, const float* __restrict__ g,
    const float* __restrict__ bta, f16* __restrict__ out) {
  int wave = threadIdx.x >> 5, lane = threadIdx.x & 31;
  int pix = blockIdx.x * 8 + wave;
  int b = pix >> 16, hw = pix & 65535;
  int h = hw >> 8, w = hw & 255;
  int hs = (h + SHIFT_) & 255, ws2 = (w + SHIFT_) & 255;
  float vals[4]; float s = 0.f, s2 = 0.f;
  #pragma unroll
  for (int j = 0; j < 4; j++) {
    int c = lane + 32 * j;
    float v = x[(((size_t)b * CDIM + c) * HH + hs) * WW + ws2];
    vals[j] = v; s += v; s2 += v * v;
  }
  s = waveSum(s); s2 = waveSum(s2);
  float mean = s * (1.f / 128.f);
  float var = s2 * (1.f / 128.f) - mean * mean;
  float inv = rsqrtf(var + 1e-5f);
  #pragma unroll
  for (int j = 0; j < 4; j++) {
    int c = lane + 32 * j;
    out[(size_t)pix * CDIM + c] = (f16)((vals[j] - mean) * inv * g[c] + bta[c]);
  }
}

// ---------------------------------------------------------------- LN2 on f16 [N,128]
__global__ __launch_bounds__(256) void ln2_kernel(
    const f16* __restrict__ yin, const float* __restrict__ g,
    const float* __restrict__ bta, f16* __restrict__ out) {
  int wave = threadIdx.x >> 5, lane = threadIdx.x & 31;
  int pix = blockIdx.x * 8 + wave;
  float vals[4]; float s = 0.f, s2 = 0.f;
  #pragma unroll
  for (int j = 0; j < 4; j++) {
    int c = lane + 32 * j;
    float v = (float)yin[(size_t)pix * CDIM + c];
    vals[j] = v; s += v; s2 += v * v;
  }
  s = waveSum(s); s2 = waveSum(s2);
  float mean = s * (1.f / 128.f);
  float var = s2 * (1.f / 128.f) - mean * mean;
  float inv = rsqrtf(var + 1e-5f);
  #pragma unroll
  for (int j = 0; j < 4; j++) {
    int c = lane + 32 * j;
    out[(size_t)pix * CDIM + c] = (f16)((vals[j] - mean) * inv * g[c] + bta[c]);
  }
}

// ---------------------------------------------------------------- generic WMMA GEMM
// out[N,O] = A[N,K](f16) @ Wt[O,K]^T(f16) + bias.  8 waves/block: 128 rows x 16 cols.
// Weight tile (16 x K f16, row-major) staged into LDS by the TDM when available.
// MODE 0: store f16 [N,O].
// MODE 2: proj epilogue: un-roll(+4,+4), add NCHW f32 shortcut, store f16 y[N,128].
// MODE 3: fc2 epilogue: add f16 residual y, store f32 NCHW to d_out.
template<int K, int MODE>
__global__ __launch_bounds__(256) void gemm_kernel(
    const f16* __restrict__ A, const f16* __restrict__ Wt,
    const float* __restrict__ bias, int O,
    f16* __restrict__ outh,
    const float* __restrict__ resx, const f16* __restrict__ resh,
    float* __restrict__ outf) {
  __shared__ f16 wlds[16 * K];
  unsigned* wl32 = (unsigned*)wlds;
  int tid = threadIdx.x;
  int oBase = blockIdx.y * 16;
  int mBase0 = blockIdx.x * 128;
#if HAVE_TDM
  if (tid < 32) {  // wave 0 issues one TDM DMA: 16 x K f16 tile -> LDS row-major
    unsigned long long ga = (unsigned long long)(const void*)(Wt + (size_t)oBase * K);
    unsigned ldsOff = (unsigned)(size_t)(void*)wlds;  // low 32 bits = LDS offset
    v4u g0;
    g0[0] = 1u;                                        // count=1, user mode
    g0[1] = ldsOff;                                    // lds_addr
    g0[2] = (unsigned)(ga & 0xffffffffu);              // global_addr[31:0]
    g0[3] = (unsigned)((ga >> 32) & 0x01ffffffu) | 0x80000000u;  // addr[56:32] | type=2
    v8i_t g1;
    g1[0] = 0x10000;                                   // data_size=1 (2 bytes)
    g1[1] = (int)((unsigned)K << 16);                  // tensor_dim0[15:0] = K
    g1[2] = (int)(16u << 16);                          // tensor_dim1[15:0] = 16
    g1[3] = (int)((unsigned)K << 16);                  // tile_dim0 = K
    g1[4] = 16;                                        // tile_dim1 = 16
    g1[5] = K;                                         // tensor_dim0_stride = K
    g1[6] = 0; g1[7] = 0;
    v4i_t gz = {0, 0, 0, 0};
    v8i_t gz8 = {0, 0, 0, 0, 0, 0, 0, 0};
    __builtin_amdgcn_tensor_load_to_lds(g0, g1, gz, gz, gz8, 0);
    __builtin_amdgcn_s_wait_tensorcnt(0);
  }
#else
  for (int i = tid; i < 8 * K; i += 256) {             // cooperative fallback
    int r = i / (K / 2); int kk = i - r * (K / 2);
    wl32[i] = ((const unsigned*)(Wt + (size_t)(oBase + r) * K))[kk];
  }
#endif
  __syncthreads();
  int wave = tid >> 5, lane = tid & 31;
  int half = lane >> 4, l16 = lane & 15;
  int mBase = mBase0 + wave * 16;
  const unsigned* arow = (const unsigned*)(A + (size_t)(mBase + l16) * K);
  __builtin_prefetch((const void*)(arow + 64 * K), 0, 1);  // next 128-row macro tile
  v8f acc = {};
  #pragma unroll
  for (int kb = 0; kb < K; kb += 32) {
    union { v16h v; unsigned u[8]; } af, bf;
    #pragma unroll
    for (int j = 0; j < 8; j++) af.u[j] = arow[(kb + a_koff(j, half)) >> 1];
    #pragma unroll
    for (int j = 0; j < 8; j++) bf.u[j] = wl32[l16 * (K / 2) + (kb >> 1) + j + 8 * half];
    acc = __builtin_amdgcn_wmma_f32_16x16x32_f16(false, af.v, false, bf.v,
                                                 (short)0, acc, false, false);
  }
  int n = oBase + l16;
  float bv = bias[n];
  #pragma unroll
  for (int j = 0; j < 8; j++) {
    int m = mBase + j + 8 * half;
    float v = acc[j] + bv;
    if (MODE == 0) {
      outh[(size_t)m * O + n] = (f16)v;
    } else if (MODE == 2) {
      int b = m >> 16, hw = m & 65535, h = hw >> 8, w = hw & 255;
      int h2 = (h + SHIFT_) & 255, w2 = (w + SHIFT_) & 255;   // un-roll
      size_t yrow = ((size_t)b << 16) + (h2 << 8) + w2;
      v += resx[(((size_t)b * CDIM + n) * HH + h2) * WW + w2];
      outh[yrow * CDIM + n] = (f16)v;
    } else {  // MODE 3
      int b = m >> 16, hw = m & 65535, h = hw >> 8, w = hw & 255;
      v += (float)resh[(size_t)m * CDIM + n];
      outf[(((size_t)b * CDIM + n) * HH + h) * WW + w] = v;
    }
  }
}

// ---------------------------------------------------------------- CASA input: xw_g (+ prev)
__global__ void casa_input_kernel(const f16* __restrict__ xwg,
                                  const f16* __restrict__ prev,
                                  f16* __restrict__ cur) {
  int i = blockIdx.x * 256 + threadIdx.x;   // over NPIX*32
  int pix = i >> 5, c = i & 31;
  float v = (float)xwg[(size_t)pix * 128 + c];
  if (prev) v += (float)prev[(size_t)pix * 128 + c];
  cur[i] = (f16)v;
}

// ---------------------------------------------------------------- depthwise conv (+cape, *scale, gelu)
__global__ __launch_bounds__(256) void dw_kernel(
    const f16* __restrict__ in, int ldin,
    const float* __restrict__ wgt, const float* __restrict__ bias,
    const f16* __restrict__ addin, float scale,
    f16* __restrict__ out, int ldout,
    int C, int kh, int kw, int dogelu) {
  int i = blockIdx.x * 256 + threadIdx.x;
  if (i >= NPIX * C) return;
  int pix = i / C, c = i - pix * C;
  int b = pix >> 16, hw = pix & 65535, h = hw >> 8, w = hw & 255;
  int ph = kh >> 1, pw = kw >> 1;
  float acc = bias[c];
  for (int dy = 0; dy < kh; dy++) {
    int hh = h + dy - ph;
    if ((unsigned)hh >= (unsigned)HH) continue;
    for (int dx = 0; dx < kw; dx++) {
      int ww = w + dx - pw;
      if ((unsigned)ww >= (unsigned)WW) continue;
      size_t p = ((size_t)b << 16) + (hh << 8) + ww;
      acc += (float)in[p * ldin + c] * wgt[c * kh * kw + dy * kw + dx];
    }
  }
  if (addin) acc += (float)addin[(size_t)pix * C + c];
  acc *= scale;
  if (dogelu) acc = 0.5f * acc * (1.f + erff(acc * 0.70710678118654752f));
  out[(size_t)pix * ldout + c] = (f16)acc;
}

// ---------------------------------------------------------------- window attention
// One workgroup (4 waves) per window. L = hs*wsz in {64,32,16}; C=32.
// V is staged TRANSPOSED (vsT[c][k], padded stride) so the P@v B-fragment is
// 8 contiguous dword LDS reads, same pattern as the q@k^T operand.
__global__ __launch_bounds__(128) void attn_kernel(
    const f16* __restrict__ qd, const f16* __restrict__ kd,
    const f16* __restrict__ vv, const f16* __restrict__ cape,
    f16* __restrict__ attout, int hs, int wsz) {
  __shared__ f16 qs[64 * 32];
  __shared__ f16 ks[64 * 32];
  __shared__ f16 vsT[32 * VSTRIDE];
  __shared__ float Ss[64 * 64];
  __shared__ f16 Ps[64 * 64];
  int L = hs * wsz;
  int Lp = L < 32 ? 32 : L;            // K-pad for WMMA when L==16
  int nw = WW / wsz;
  int winPerImg = (HH / hs) * nw;
  int wi = blockIdx.x;
  int b = wi / winPerImg, r = wi - b * winPerImg;
  int whi = r / nw, wwi = r - whi * nw;
  int tid = threadIdx.x;

  for (int i = tid; i < L * 16; i += 128) {          // stage q,k (row-major), v (transposed)
    int l = i >> 4, c2 = i & 15;
    int hy = l / wsz, wx = l - hy * wsz;
    size_t grow = ((size_t)b << 16) + ((whi * hs + hy) << 8) + (wwi * wsz + wx);
    ((unsigned*)qs)[i] = ((const unsigned*)(qd + grow * 32))[c2];
    ((unsigned*)ks)[i] = ((const unsigned*)(kd + grow * 32))[c2];
    union { unsigned u; f16 hx[2]; } pv;
    pv.u = ((const unsigned*)(vv + grow * 32))[c2];
    vsT[(2 * c2 + 0) * VSTRIDE + l] = pv.hx[0];
    vsT[(2 * c2 + 1) * VSTRIDE + l] = pv.hx[1];
  }
  if (L < 32) {                                      // zero K-pad columns of vsT
    for (int i = tid; i < 32 * (32 - L); i += 128) {
      int c = i / (32 - L), l = L + (i - c * (32 - L));
      vsT[c * VSTRIDE + l] = (f16)0.f;
    }
  }
  __syncthreads();

  int wave = tid >> 5, lane = tid & 31;
  int half = lane >> 4, l16 = lane & 15;
  int nT = L >> 4;

  // S = q @ k^T   (K=32, one WMMA per 16x16 tile)
  for (int t = wave; t < nT * nT; t += 4) {
    int tm = t / nT, tn = t - tm * nT;
    union { v16h v; unsigned u[8]; } af, bf;
    const unsigned* aq = (const unsigned*)qs + (tm * 16 + l16) * 16;
    const unsigned* bk = (const unsigned*)ks + (tn * 16 + l16) * 16;
    #pragma unroll
    for (int j = 0; j < 8; j++) af.u[j] = aq[a_koff(j, half) >> 1];
    #pragma unroll
    for (int j = 0; j < 8; j++) bf.u[j] = bk[j + 8 * half];
    v8f acc = {};
    acc = __builtin_amdgcn_wmma_f32_16x16x32_f16(false, af.v, false, bf.v,
                                                 (short)0, acc, false, false);
    #pragma unroll
    for (int j = 0; j < 8; j++)
      Ss[(tm * 16 + j + 8 * half) * Lp + tn * 16 + l16] = acc[j];
  }
  __syncthreads();

  // row softmax -> Ps (f16), zero the K-pad columns
  for (int rr = tid; rr < L; rr += 128) {
    float mx = -3.4e38f;
    for (int c = 0; c < L; c++) mx = fmaxf(mx, Ss[rr * Lp + c]);
    float sum = 0.f;
    for (int c = 0; c < L; c++) { float e = __expf(Ss[rr * Lp + c] - mx); Ss[rr * Lp + c] = e; sum += e; }
    float inv = 1.f / sum;
    for (int c = 0; c < L; c++) Ps[rr * Lp + c] = (f16)(Ss[rr * Lp + c] * inv);
    for (int c = L; c < Lp; c++) Ps[rr * Lp + c] = (f16)0.f;
  }
  __syncthreads();

  // O = P @ v  (Mtiles=nT, Ntiles=2, K=Lp) ; fused "+ cape" on store
  for (int t = wave; t < nT * 2; t += 4) {
    int tm = t >> 1, tn = t & 1;
    v8f acc = {};
    const unsigned* bvp = (const unsigned*)vsT + (tn * 16 + l16) * (VSTRIDE / 2);
    for (int kb = 0; kb < Lp; kb += 32) {
      union { v16h v; unsigned u[8]; } af, bf;
      const unsigned* ap = (const unsigned*)Ps + (((tm * 16 + l16) * Lp) >> 1);
      #pragma unroll
      for (int j = 0; j < 8; j++) af.u[j] = ap[(kb + a_koff(j, half)) >> 1];
      #pragma unroll
      for (int j = 0; j < 8; j++) bf.u[j] = bvp[(kb >> 1) + j + 8 * half];
      acc = __builtin_amdgcn_wmma_f32_16x16x32_f16(false, af.v, false, bf.v,
                                                   (short)0, acc, false, false);
    }
    int n = tn * 16 + l16;
    #pragma unroll
    for (int j = 0; j < 8; j++) {
      int l = tm * 16 + j + 8 * half;
      int hy = l / wsz, wx = l - hy * wsz;
      size_t grow = ((size_t)b << 16) + ((whi * hs + hy) << 8) + (wwi * wsz + wx);
      attout[grow * 128 + n] = (f16)(acc[j] + (float)cape[grow * 32 + n]);
    }
  }
}

// ================================================================ host
extern "C" void kernel_launch(void* const* d_in, const int* in_sizes, int n_in,
                              void* d_out, int out_size, void* d_ws, size_t ws_size,
                              hipStream_t stream) {
  (void)in_sizes; (void)n_in; (void)out_size; (void)ws_size;
  const float* x      = (const float*)d_in[0];
  const float* n1_g   = (const float*)d_in[1];
  const float* n1_b   = (const float*)d_in[2];
  const float* n2_g   = (const float*)d_in[3];
  const float* n2_b   = (const float*)d_in[4];
  const float* p1_w   = (const float*)d_in[5];
  const float* p1_b   = (const float*)d_in[6];
  const float* proj_w = (const float*)d_in[7];
  const float* proj_b = (const float*)d_in[8];
  const float* fc1_w  = (const float*)d_in[49];
  const float* fc1_b  = (const float*)d_in[50];
  const float* fc2_w  = (const float*)d_in[51];
  const float* fc2_b  = (const float*)d_in[52];

  char* ws = (char*)d_ws;
  // region map (bytes); overlays are only reused after producer region is dead
  f16* sn   = (f16*)(ws + 0);               // [N,128] f16   (later: y)
  f16* xw   = (f16*)(ws + 67108864);        // [N,128] f16   (later: z)
  f16* att  = (f16*)(ws + 134217728);       // [N,128] f16   (later: part of h)
  char* scr = ws + 201326592;               // CASA scratch (dead before MLP)
  f16* cur  = (f16*)scr;                    // [N,32]
  f16* qkb  = (f16*)(scr + 16777216);       // [N,64]
  f16* vb   = (f16*)(scr + 50331648);       // [N,32]
  f16* qd   = (f16*)(scr + 67108864);       // [N,32]
  f16* kd   = (f16*)(scr + 83886080);       // [N,32]
  f16* cape = (f16*)(scr + 100663296);      // [N,32]
  f16* y    = (f16*)(ws + 0);               // [N,128] f16 (overlays sn)
  f16* z    = (f16*)(ws + 67108864);        // [N,128] f16 (overlays xw)
  f16* h    = (f16*)(ws + 134217728);       // [N,256] f16 (overlays att+scr head)
  f16* dwt  = (f16*)(ws + 268435456);       // [N,64]  f16
  f16* h2   = (f16*)(ws + 301989888);       // [N,256] f16
  f16* wa   = (f16*)(ws + 436207616);       // f16 weight arena (~221KB)

  f16* wp1  = wa;                           // 16384
  f16* wqk0 = wa + 16384;                   // 4 x 2048
  f16* wv0  = wa + 24576;                   // 4 x 1024
  f16* wproj= wa + 28672;                   // 16384
  f16* wfc1 = wa + 45056;                   // 32768
  f16* wfc2 = wa + 77824;                   // 32768

  auto pack = [&](const float* s, f16* d, int n) {
    packw_kernel<<<(n + 255) / 256, 256, 0, stream>>>(s, d, n);
  };
  pack(p1_w, wp1, 16384);
  for (int g = 0; g < 4; g++) {
    pack((const float*)d_in[9 + g * 10 + 0], wqk0 + g * 2048, 2048);
    pack((const float*)d_in[9 + g * 10 + 6], wv0 + g * 1024, 1024);
  }
  pack(proj_w, wproj, 16384);
  pack(fc1_w, wfc1, 32768);
  pack(fc2_w, wfc2, 32768);

  // LN1 (with roll -4,-4)
  ln_shift_kernel<<<NPIX / 8, 256, 0, stream>>>(x, n1_g, n1_b, sn);

  // xw = conv1x1(sn, p1)
  gemm_kernel<128, 0><<<dim3(NPIX / 128, 8), 256, 0, stream>>>(
      sn, wp1, p1_b, 128, xw, nullptr, nullptr, nullptr);

  const int win_h[4] = {8, 8, 4, 4};
  const int win_w[4] = {8, 4, 8, 4};
  int nblk32 = NPIX * 32 / 256;
  for (int g = 0; g < 4; g++) {
    const float* qk_b_  = (const float*)d_in[9 + g * 10 + 1];
    const float* q5_w   = (const float*)d_in[9 + g * 10 + 2];
    const float* q5_b   = (const float*)d_in[9 + g * 10 + 3];
    const float* k5_w   = (const float*)d_in[9 + g * 10 + 4];
    const float* k5_b   = (const float*)d_in[9 + g * 10 + 5];
    const float* v_b_   = (const float*)d_in[9 + g * 10 + 7];
    const float* cape_w = (const float*)d_in[9 + g * 10 + 8];
    const float* cape_b = (const float*)d_in[9 + g * 10 + 9];

    casa_input_kernel<<<nblk32, 256, 0, stream>>>(
        xw + g * 32, g ? att + (g - 1) * 32 : nullptr, cur);
    gemm_kernel<32, 0><<<dim3(NPIX / 128, 4), 256, 0, stream>>>(
        cur, wqk0 + g * 2048, qk_b_, 64, qkb, nullptr, nullptr, nullptr);
    gemm_kernel<32, 0><<<dim3(NPIX / 128, 2), 256, 0, stream>>>(
        cur, wv0 + g * 1024, v_b_, 32, vb, nullptr, nullptr, nullptr);
    dw_kernel<<<nblk32, 256, 0, stream>>>(vb, 32, cape_w, cape_b, nullptr, 1.f,
                                          cape, 32, 32, 5, 5, 0);
    dw_kernel<<<nblk32, 256, 0, stream>>>(qkb, 64, q5_w, q5_b, cape, SCALE_Q,
                                          qd, 32, 32, 5, 5, 0);
    dw_kernel<<<nblk32, 256, 0, stream>>>(qkb + 32, 64, k5_w, k5_b, cape, 1.f,
                                          kd, 32, 32, 5, 5, 0);
    int nwin = BB * (HH / win_h[g]) * (WW / win_w[g]);
    attn_kernel<<<nwin, 128, 0, stream>>>(qd, kd, vb, cape, att + g * 32,
                                          win_h[g], win_w[g]);
  }

  // y = unroll(att @ proj^T + b) + shortcut(x)
  gemm_kernel<128, 2><<<dim3(NPIX / 128, 8), 256, 0, stream>>>(
      att, wproj, proj_b, 128, y, x, nullptr, nullptr);

  // z = LN2(y)
  ln2_kernel<<<NPIX / 8, 256, 0, stream>>>(y, n2_g, n2_b, z);

  // h = fc1(z)
  gemm_kernel<128, 0><<<dim3(NPIX / 128, 16), 256, 0, stream>>>(
      z, wfc1, fc1_b, 256, h, nullptr, nullptr, nullptr);

  // depthwise branches + GELU -> h2
  int nblk64 = NPIX * 64 / 256;
  dw_kernel<<<nblk64, 256, 0, stream>>>(h,       256, (const float*)d_in[53], (const float*)d_in[54], nullptr, 1.f, h2,       256, 64, 3, 3, 1);
  dw_kernel<<<nblk64, 256, 0, stream>>>(h + 64,  256, (const float*)d_in[55], (const float*)d_in[56], nullptr, 1.f, h2 + 64,  256, 64, 5, 3, 1);
  dw_kernel<<<nblk64, 256, 0, stream>>>(h + 128, 256, (const float*)d_in[57], (const float*)d_in[58], nullptr, 1.f, h2 + 128, 256, 64, 3, 5, 1);
  dw_kernel<<<nblk64, 256, 0, stream>>>(h + 192, 256, (const float*)d_in[59], (const float*)d_in[60], nullptr, 1.f, dwt,      64,  64, 3, 3, 0);
  dw_kernel<<<nblk64, 256, 0, stream>>>(dwt,     64,  (const float*)d_in[61], (const float*)d_in[62], nullptr, 1.f, h2 + 192, 256, 64, 3, 3, 1);

  // out = NCHW( y + fc2(h2) )
  gemm_kernel<256, 3><<<dim3(NPIX / 128, 8), 256, 0, stream>>>(
      h2, wfc2, fc2_b, 128, nullptr, nullptr, y, (float*)d_out);
}
